// SVDQuantLinearManual_48773648613649
// MI455X (gfx1250) — 8-bit matrix operand order repaired and checked by reading, compiled, not run on hardware
//
#include <hip/hip_runtime.h>
#include <hip/hip_bf16.h>

typedef int   v8i __attribute__((ext_vector_type(8)));
typedef int   v4i __attribute__((ext_vector_type(4)));
typedef int   v2i __attribute__((ext_vector_type(2)));
typedef float v8f __attribute__((ext_vector_type(8)));
typedef float v4f __attribute__((ext_vector_type(4)));

#define B_TOK 64
#define K_DIM 4096
#define N_DIM 12288
#define R_DIM 32
#define NGRP  64
#define MT    4

__global__ __launch_bounds__(32) void k_quant_pack(
    const float* __restrict__ x, const float* __restrict__ smooth,
    int* __restrict__ Afrag, float* __restrict__ ASfrag) {
  const int g     = blockIdx.x;
  const int mt    = blockIdx.y;
  const int lane  = threadIdx.x;
  const int half  = lane >> 4;
  const int row   = lane & 15;
  const int token = mt * 16 + row;

  float vals[8][4];
  float amax = 0.0f;
#pragma unroll
  for (int v = 0; v < 8; ++v) {
    const int kb = g * 64 + (v >> 1) * 16 + (v & 1) * 4 + half * 8;
    const v4f xs = *(const v4f*)(x + (size_t)token * K_DIM + kb);
    const v4f sm = *(const v4f*)(smooth + kb);
#pragma unroll
    for (int j = 0; j < 4; ++j) {
      const float xv = xs[j] / sm[j];
      vals[v][j] = xv;
      amax = fmaxf(amax, fabsf(xv));
    }
  }
  amax = fmaxf(amax, __shfl_xor(amax, 16, 32));
  const float ascale = fmaxf(amax * (1.0f / 7.0f), 1e-8f);
  const float inv    = 1.0f / ascale;

  const int base = ((mt * NGRP + g) * 32 + lane) * 8;
  v4i lo, hi;
#pragma unroll
  for (int v = 0; v < 8; ++v) {
    int d = 0;
#pragma unroll
    for (int j = 0; j < 4; ++j) {
      int q = (int)rintf(vals[v][j] * inv);
      q = q < -8 ? -8 : (q > 7 ? 7 : q);
      d |= (q & 0xFF) << (8 * j);
    }
    if (v < 4) lo[v] = d; else hi[v - 4] = d;
  }
  *(v4i*)(Afrag + base)     = lo;
  *(v4i*)(Afrag + base + 4) = hi;

  v4f slo, shi;
#pragma unroll
  for (int r = 0; r < 8; ++r) {
    const float ar = __shfl(ascale, r + 8 * half, 32);
    if (r < 4) slo[r] = ar; else shi[r - 4] = ar;
  }
  *(v4f*)(ASfrag + base)     = slo;
  *(v4f*)(ASfrag + base + 4) = shi;
}

__global__ __launch_bounds__(256) void k_lora_down(
    const float* __restrict__ x, const float* __restrict__ smooth,
    const float* __restrict__ ld, float* __restrict__ t) {
  __shared__ float red[256];
  const int b     = blockIdx.x;
  const int tid   = threadIdx.x;
  const int r     = tid & 31;
  const int slice = tid >> 5;
  float p = 0.0f;
  const int k0 = slice * 512;
  for (int k = k0; k < k0 + 512; ++k)
    p += (x[(size_t)b * K_DIM + k] / smooth[k]) * ld[(size_t)k * R_DIM + r];
  red[tid] = p;
  __syncthreads();
  if (tid < 32) {
    float s = 0.0f;
#pragma unroll
    for (int i = 0; i < 8; ++i) s += red[tid + i * 32];
    t[b * R_DIM + tid] = s;
  }
}

__global__ __launch_bounds__(256) void k_lora_up(
    const float* __restrict__ t, const float* __restrict__ lup,
    const float* __restrict__ bias, float* __restrict__ out) {
  __shared__ float ts[R_DIM];
  const int b = blockIdx.y;
  const int n = blockIdx.x * 256 + threadIdx.x;
  if (threadIdx.x < R_DIM) ts[threadIdx.x] = t[b * R_DIM + threadIdx.x];
  __syncthreads();
  float acc = bias[n];
  const v4f* lrow = (const v4f*)(lup + (size_t)n * R_DIM);
#pragma unroll
  for (int j = 0; j < 8; ++j) {
    const v4f l = lrow[j];
#pragma unroll
    for (int c = 0; c < 4; ++c) acc += ts[4 * j + c] * l[c];
  }
  out[(size_t)b * N_DIM + n] = acc;
}

__global__ __launch_bounds__(32 * MT) void k_gemm(
    const int* __restrict__ qw, const float* __restrict__ wsc,
    const int* __restrict__ Afrag, const float* __restrict__ ASfrag,
    float* __restrict__ out) {
  __shared__ int ldsB[2][32 * 8];
  const int lane = threadIdx.x & 31;
  const int mt   = threadIdx.x >> 5;
  const int nt   = blockIdx.x;
  const int half = lane >> 4;
  const int col  = lane & 15;
  const int n    = nt * 16 + col;
  const int* __restrict__ wrow = qw + (size_t)n * K_DIM;

  float acc[8];
#pragma unroll
  for (int r = 0; r < 8; ++r) acc[r] = 0.0f;

  const v8i czero = {0, 0, 0, 0, 0, 0, 0, 0};

  for (int g = 0; g < NGRP; ++g) {
    v2i pk;
#pragma unroll
    for (int s = 0; s < 2; ++s) {
      const int v  = 2 * mt + s;
      const int kb = g * 64 + (v >> 1) * 16 + (v & 1) * 4 + half * 8;
      const v4i w  = *(const v4i*)(wrow + kb);
      const unsigned p01 =
          __builtin_amdgcn_perm((unsigned)w[1], (unsigned)w[0], 0x0C0C0400u);
      const unsigned p23 =
          __builtin_amdgcn_perm((unsigned)w[3], (unsigned)w[2], 0x0C0C0400u);
      pk[s] = (int)__builtin_amdgcn_perm(p23, p01, 0x05040100u);
    }
    *(v2i*)&ldsB[g & 1][lane * 8 + 2 * mt] = pk;

    if (g + 1 < NGRP)
      __builtin_prefetch(wrow + (g + 1) * 64, 0, 0);

    __syncthreads();

    const float wsv  = wsc[(size_t)g * N_DIM + n];
    const v8i bfrag  = *(const v8i*)&ldsB[g & 1][lane * 8];
    const int base   = ((mt * NGRP + g) * 32 + lane) * 8;
    const v8i afrag  = *(const v8i*)(Afrag + base);
    const v8f asf    = *(const v8f*)(ASfrag + base);
    const v8i d = __builtin_amdgcn_wmma_i32_16x16x64_iu8(
         true, afrag,  true, bfrag, czero,
         false,  false);
#pragma unroll
    for (int r = 0; r < 8; ++r)
      acc[r] += (float)d[r] * (asf[r] * wsv);
  }

#pragma unroll
  for (int r = 0; r < 8; ++r) {
    const int token  = mt * 16 + r + 8 * half;
    const size_t idx = (size_t)token * N_DIM + n;
    out[idx] += acc[r];
  }
}

extern "C" void kernel_launch(void* const* d_in, const int* in_sizes, int n_in,
                              void* d_out, int out_size, void* d_ws, size_t ws_size,
                              hipStream_t stream) {
  const float* x      = (const float*)d_in[0];
  const int*   qw     = (const int*)  d_in[1];
  const float* wsc    = (const float*)d_in[2];
  const float* ldown  = (const float*)d_in[3];
  const float* lup    = (const float*)d_in[4];
  const float* smooth = (const float*)d_in[5];
  const float* bias   = (const float*)d_in[6];
  float* out = (float*)d_out;

  int*   Afrag  = (int*)d_ws;
  float* ASfrag = (float*)((char*)d_ws + 262144);
  float* t      = (float*)((char*)d_ws + 524288);

  k_quant_pack<<<dim3(NGRP, MT), 32, 0, stream>>>(x, smooth, Afrag, ASfrag);
  k_lora_down<<<dim3(B_TOK), 256, 0, stream>>>(x, smooth, ldown, t);
  k_lora_up<<<dim3(N_DIM / 256, B_TOK), 256, 0, stream>>>(t, lup, bias, out);
  k_gemm<<<dim3(N_DIM / 16), 32 * MT, 0, stream>>>(qw, wsc, Afrag, ASfrag, out);
}
